// ViT_88527865905337
// MI455X (gfx1250) — matrix operand order repaired and checked by reading, not compiled, not run on hardware
//
#include <hip/hip_runtime.h>
#include <math.h>

// ---------------- model constants ----------------
#define DD   768
#define HH   12
#define HDIM 64
#define MLPD 3072
#define BB   32
#define NP   196
#define NMAX 197
#define NSMAX 224   // NMAX rounded up to 32

typedef __attribute__((ext_vector_type(16))) _Float16 v16h;
typedef __attribute__((ext_vector_type(8)))  _Float16 v8h;
typedef __attribute__((ext_vector_type(8)))  float    v8f;
typedef __attribute__((ext_vector_type(4)))  float    v4f;

__device__ __forceinline__ v8f wmma32(v16h a, v16h b, v8f c) {
  // D = A(16x32 f16) * B(32x16 f16) + C(16x16 f32)
  return __builtin_amdgcn_wmma_f32_16x16x32_f16(false, a, false, b, (short)0, c,
                                                false, false);
}

// ---------------- weight convert: f32 [K][N] -> f16 [N][K] ----------------
__global__ void convw_kernel(const float* __restrict__ W, _Float16* __restrict__ Wt,
                             int K, int N) {
  int i = blockIdx.x * 256 + threadIdx.x;
  if (i >= K * N) return;
  int k = i % K, n = i / K;
  Wt[i] = (_Float16)W[(size_t)k * N + n];
}
// direct convert (already [N][K] in memory, e.g. patch_w [Dout][768])
__global__ void convd_kernel(const float* __restrict__ W, _Float16* __restrict__ Wt,
                             int total) {
  int i = blockIdx.x * 256 + threadIdx.x;
  if (i >= total) return;
  Wt[i] = (_Float16)W[i];
}

// ---------------- LayerNorm (fp32 in -> f16 out) ----------------
__global__ __launch_bounds__(256) void ln_kernel(const float* __restrict__ x,
                                                 const float* __restrict__ g,
                                                 const float* __restrict__ b,
                                                 _Float16* __restrict__ out) {
  __shared__ float red[256];
  size_t row = blockIdx.x;
  const float* xr = x + row * DD;
  float s = 0.f;
  for (int c = threadIdx.x; c < DD; c += 256) s += xr[c];
  red[threadIdx.x] = s; __syncthreads();
  for (int st = 128; st > 0; st >>= 1) {
    if (threadIdx.x < st) red[threadIdx.x] += red[threadIdx.x + st];
    __syncthreads();
  }
  float mean = red[0] / DD; __syncthreads();
  float vs = 0.f;
  for (int c = threadIdx.x; c < DD; c += 256) { float d = xr[c] - mean; vs += d * d; }
  red[threadIdx.x] = vs; __syncthreads();
  for (int st = 128; st > 0; st >>= 1) {
    if (threadIdx.x < st) red[threadIdx.x] += red[threadIdx.x + st];
    __syncthreads();
  }
  float rstd = rsqrtf(red[0] / DD + 1e-5f);
  for (int c = threadIdx.x; c < DD; c += 256)
    out[row * DD + c] = (_Float16)((xr[c] - mean) * rstd * g[c] + b[c]);
}

// ---------------- im2col for 16x16/s16 patch conv ----------------
__global__ void im2col_kernel(const float* __restrict__ x, _Float16* __restrict__ A,
                              int total) {
  int i = blockIdx.x * 256 + threadIdx.x;
  if (i >= total) return;
  int k  = i % DD;            // c*256 + py*16 + px
  int pi = (i / DD) % NP;
  int b  = i / (DD * NP);
  int c = k >> 8, py = (k >> 4) & 15, px = k & 15;
  int gy = pi / 14, gx = pi % 14;
  A[i] = (_Float16)x[(((size_t)b * 3 + c) * 224 + gy * 16 + py) * 224 + gx * 16 + px];
}

// ---------------- generic WMMA GEMM ----------------
// C[M,N] = epilogue(A[f16,MxK,row-major] * Wt[f16,NxK i.e. B^T row-major])
// 4 waves / block, 16x64 output tile, K % 32 == 0.
// Out-of-range rows/cols are CLAMPED (their D entries are never stored).
__global__ __launch_bounds__(128) void gemm_kernel(
    const _Float16* __restrict__ A, int lda,
    const _Float16* __restrict__ Wt,
    const float* __restrict__ bias,
    const float* __restrict__ resid, int ldr,
    float* __restrict__ Cf, _Float16* __restrict__ Ch, int ldc,
    int M, int N, int K, int gelu) {
  int lane = threadIdx.x & 31, wave = threadIdx.x >> 5;
  int m0 = blockIdx.x * 16;
  int n0 = blockIdx.y * 64 + wave * 16;
  if (n0 >= N) return;                       // wave-uniform: EXEC stays full
  int rl = lane & 15, g = lane >> 4;
  int ar = m0 + rl; if (ar > M - 1) ar = M - 1;
  int bc = n0 + rl; if (bc > N - 1) bc = N - 1;
  const _Float16* Ap = A  + (size_t)ar * lda + (g << 3);   // +8g
  const _Float16* Bp = Wt + (size_t)bc * K   + (g << 3);   // +8g
  v8f acc = {};
  for (int k0 = 0; k0 < K; k0 += 32) {
    v8h a0 = *(const v8h*)(Ap + k0);
    v8h a1 = *(const v8h*)(Ap + k0 + 16);
    v8h b0 = *(const v8h*)(Bp + k0);
    v8h b1 = *(const v8h*)(Bp + k0 + 16);
    v16h af = __builtin_shufflevector(a0, a1, 0,1,2,3,4,5,6,7,8,9,10,11,12,13,14,15);
    v16h bf = __builtin_shufflevector(b0, b1, 0,1,2,3,4,5,6,7,8,9,10,11,12,13,14,15);
    acc = wmma32(af, bf, acc);
  }
#pragma unroll
  for (int r = 0; r < 8; ++r) {
    int row = m0 + r + (g << 3), col = n0 + rl;
    if (row < M && col < N) {
      float v = acc[r];
      if (bias)  v += bias[col];
      if (gelu)  v = 0.5f * v * (1.f + erff(v * 0.70710678118654752f));
      if (resid) v += resid[(size_t)row * ldr + col];
      if (Cf) Cf[(size_t)row * ldc + col] = v;
      if (Ch) Ch[(size_t)row * ldc + col] = (_Float16)v;
    }
  }
}

// ---------------- QK^T per head (scale folded into Q) ----------------
__global__ __launch_bounds__(128) void qk_kernel(const float* __restrict__ qkv,
                                                 float* __restrict__ logits,
                                                 int N, int NS) {
  int lane = threadIdx.x & 31, wave = threadIdx.x >> 5;
  int bh = blockIdx.z, b = bh / HH, h = bh % HH;
  int m0 = blockIdx.x * 16;
  int n0 = blockIdx.y * 64 + wave * 16;
  if (n0 >= N) return;
  int rl = lane & 15, g = lane >> 4;
  int qi = m0 + rl; if (qi > N - 1) qi = N - 1;
  int kj = n0 + rl; if (kj > N - 1) kj = N - 1;
  const float* qr = qkv + (size_t)(b * N + qi) * (3 * DD) + h * HDIM + (g << 3);
  const float* kr = qkv + (size_t)(b * N + kj) * (3 * DD) + DD + h * HDIM + (g << 3);
  v8f acc = {};
#pragma unroll
  for (int k0 = 0; k0 < HDIM; k0 += 32) {
    v4f a0 = *(const v4f*)(qr + k0);
    v4f a1 = *(const v4f*)(qr + k0 + 4);
    v4f a2 = *(const v4f*)(qr + k0 + 16);
    v4f a3 = *(const v4f*)(qr + k0 + 20);
    v4f c0 = *(const v4f*)(kr + k0);
    v4f c1 = *(const v4f*)(kr + k0 + 4);
    v4f c2 = *(const v4f*)(kr + k0 + 16);
    v4f c3 = *(const v4f*)(kr + k0 + 20);
    v16h af, bf;
#pragma unroll
    for (int i = 0; i < 4; ++i) {
      af[i]      = (_Float16)(a0[i] * 0.125f);
      af[i + 4]  = (_Float16)(a1[i] * 0.125f);
      af[i + 8]  = (_Float16)(a2[i] * 0.125f);
      af[i + 12] = (_Float16)(a3[i] * 0.125f);
      bf[i]      = (_Float16)c0[i];
      bf[i + 4]  = (_Float16)c1[i];
      bf[i + 8]  = (_Float16)c2[i];
      bf[i + 12] = (_Float16)c3[i];
    }
    acc = wmma32(af, bf, acc);
  }
#pragma unroll
  for (int r = 0; r < 8; ++r) {
    int row = m0 + r + (g << 3), col = n0 + rl;
    if (row < N && col < N)
      logits[((size_t)bh * N + row) * NS + col] = acc[r];
  }
}

// ---------------- softmax in place (fp32), zero-fill column padding ----------------
__global__ __launch_bounds__(256) void softmax_kernel(float* __restrict__ att,
                                                      int N, int NS) {
  __shared__ float red[256];
  size_t row = blockIdx.x;
  float* a = att + row * NS;
  float m = -1e30f;
  for (int j = threadIdx.x; j < N; j += 256) m = fmaxf(m, a[j]);
  red[threadIdx.x] = m; __syncthreads();
  for (int st = 128; st > 0; st >>= 1) {
    if (threadIdx.x < st) red[threadIdx.x] = fmaxf(red[threadIdx.x], red[threadIdx.x + st]);
    __syncthreads();
  }
  m = red[0]; __syncthreads();
  float s = 0.f;
  for (int j = threadIdx.x; j < N; j += 256) { float e = __expf(a[j] - m); a[j] = e; s += e; }
  red[threadIdx.x] = s; __syncthreads();
  for (int st = 128; st > 0; st >>= 1) {
    if (threadIdx.x < st) red[threadIdx.x] += red[threadIdx.x + st];
    __syncthreads();
  }
  float inv = 1.f / red[0];
  for (int j = threadIdx.x; j < N; j += 256) a[j] *= inv;
  for (int j = N + threadIdx.x; j < NS; j += 256) a[j] = 0.f;  // pad for attn@V
}

// ---------------- attn @ V per head -> f16 msa [B,N,768] ----------------
__global__ __launch_bounds__(128) void av_kernel(const float* __restrict__ att,
                                                 const float* __restrict__ qkv,
                                                 _Float16* __restrict__ msa,
                                                 int N, int NS) {
  int lane = threadIdx.x & 31, wave = threadIdx.x >> 5;
  int bh = blockIdx.z, b = bh / HH, h = bh % HH;
  int m0 = blockIdx.x * 16;
  int n0 = wave * 16;                         // 0..63 within head dim
  int rl = lane & 15, g = lane >> 4;
  int ai = m0 + rl; if (ai > N - 1) ai = N - 1;
  int dcol = n0 + rl;
  const float* ar = att + ((size_t)bh * N + ai) * NS + (g << 3);
  const float* vb = qkv + 2 * DD + h * HDIM + dcol;
  v8f acc = {};
  for (int k0 = 0; k0 < NS; k0 += 32) {        // NS % 32 == 0; padding probs are 0
    v4f a0 = *(const v4f*)(ar + k0);
    v4f a1 = *(const v4f*)(ar + k0 + 4);
    v4f a2 = *(const v4f*)(ar + k0 + 16);
    v4f a3 = *(const v4f*)(ar + k0 + 20);
    v16h af, bf;
#pragma unroll
    for (int i = 0; i < 4; ++i) {
      af[i]      = (_Float16)a0[i];
      af[i + 4]  = (_Float16)a1[i];
      af[i + 8]  = (_Float16)a2[i];
      af[i + 12] = (_Float16)a3[i];
    }
#pragma unroll
    for (int i = 0; i < 16; ++i) {
      int kb = k0 + (g << 3) + (i < 8 ? i : 16 + (i - 8));
      if (kb > N - 1) kb = N - 1;             // prob is 0 there, value irrelevant
      bf[i] = (_Float16)vb[(size_t)(b * N + kb) * (3 * DD)];
    }
    acc = wmma32(af, bf, acc);
  }
#pragma unroll
  for (int r = 0; r < 8; ++r) {
    int row = m0 + r + (g << 3);
    if (row < N)
      msa[(size_t)(b * N + row) * DD + h * HDIM + n0 + rl] = (_Float16)acc[r];
  }
}

// ---------------- cls/pos assembly ----------------
__global__ void assemble_kernel(const float* __restrict__ xe, const float* __restrict__ cls,
                                const float* __restrict__ pos, float* __restrict__ h,
                                int total) {
  int i = blockIdx.x * 256 + threadIdx.x;
  if (i >= total) return;
  int c = i % DD, t = (i / DD) % NMAX, b = i / (DD * NMAX);
  float base = (t == 0) ? cls[c] : xe[((size_t)b * NP + (t - 1)) * DD + c];
  h[i] = base + pos[t * DD + c];
}

// ---------------- selector: attention stats ----------------
__global__ __launch_bounds__(256) void colmean_kernel(const float* __restrict__ att,
                                                      float* __restrict__ cm,
                                                      float* __restrict__ c0,
                                                      int N, int NS) {
  int bh = blockIdx.x, n = N - 1;
  const float* a = att + (size_t)bh * N * NS;
  for (int j = threadIdx.x; j < n; j += 256) {
    float s = 0.f;
    for (int i = 0; i < n; ++i) s += a[(size_t)(1 + i) * NS + 1 + j];
    cm[(size_t)bh * NP + j] = s / n;
  }
  if (threadIdx.x == 0) {
    float s = 0.f;
    for (int i = 0; i < n; ++i) s += a[(size_t)(1 + i) * NS];
    c0[bh] = s / n;
  }
}

__global__ __launch_bounds__(256) void tokenattn_kernel(const float* __restrict__ att,
                                                        const float* __restrict__ cm,
                                                        const float* __restrict__ c0,
                                                        float* __restrict__ ta,
                                                        int N, int NS) {
  int b = blockIdx.x, n = N - 1;
  for (int j = threadIdx.x; j < n; j += 256) {
    float cs = 0.f, is = 0.f;
    for (int h = 0; h < HH; ++h) {
      int bh = b * HH + h;
      const float* a = att + (size_t)bh * N * NS;
      cs += a[1 + j] * c0[bh];               // attn[b,h,0,1+j] * mean_i attn[...,i,0]
      is += cm[(size_t)bh * NP + j] * a[0];  // colmean * attn[b,h,0,0]
    }
    ta[b * NP + j] = (cs + is) / HH;
  }
}

// ---------------- selector: g = ifft(gaussian highpass kernel) ----------------
__global__ void gfilter_kernel(float* __restrict__ gre, float* __restrict__ gim,
                               int n, float cutoff) {
  int t = blockIdx.x * 256 + threadIdx.x;
  if (t >= n) return;
  float d0 = n * cutoff * 0.5f; d0 *= d0;
  float m0 = (n - 1) * 0.5f;
  float re = 0.f, im = 0.f;
  for (int f = 0; f < n; ++f) {
    float df = f - m0;
    float kern = 1.f - __expf(-(df * df) / (2.f * d0));
    float ang = 6.28318530717958647f * (float)f * (float)t / (float)n;
    re += kern * __cosf(ang);
    im += kern * __sinf(ang);
  }
  gre[t] = re / n; gim[t] = im / n;
}

// ---------------- selector: |ifft(K*fft(x))| via circular conv ----------------
__global__ __launch_bounds__(256) void lfe_kernel(const float* __restrict__ h,
                                                  const float* __restrict__ gre,
                                                  const float* __restrict__ gim,
                                                  float* __restrict__ lfe, int N) {
  __shared__ float red[256];
  int n = N - 1;
  int j = blockIdx.x % n, b = blockIdx.x / n;
  float s = 0.f;
  for (int c = threadIdx.x; c < DD; c += 256) {
    float re = 0.f, im = 0.f;
    int d = j;
    for (int m = 0; m < n; ++m) {
      float xv = h[((size_t)b * N + 1 + m) * DD + c];
      re += xv * gre[d]; im += xv * gim[d];
      if (--d < 0) d += n;
    }
    s += sqrtf(re * re + im * im);
  }
  red[threadIdx.x] = s; __syncthreads();
  for (int st = 128; st > 0; st >>= 1) {
    if (threadIdx.x < st) red[threadIdx.x] += red[threadIdx.x + st];
    __syncthreads();
  }
  if (threadIdx.x == 0) lfe[b * NP + j] = red[0];
}

__global__ __launch_bounds__(256) void denom_kernel(const float* __restrict__ h,
                                                    float* __restrict__ den, int N) {
  __shared__ float red[256];
  int b = blockIdx.x, n = N - 1;
  const float* base = h + ((size_t)b * N + 1) * DD;
  float s = 0.f;
  int tot = n * DD;
  for (int i = threadIdx.x; i < tot; i += 256) s += fabsf(base[i]);
  red[threadIdx.x] = s; __syncthreads();
  for (int st = 128; st > 0; st >>= 1) {
    if (threadIdx.x < st) red[threadIdx.x] += red[threadIdx.x + st];
    __syncthreads();
  }
  if (threadIdx.x == 0) den[b] = red[0] / n;
}

__global__ void scores_kernel(const float* __restrict__ lfe, const float* __restrict__ den,
                              const float* __restrict__ ta, float* __restrict__ sc, int n) {
  int i = blockIdx.x * 256 + threadIdx.x;
  if (i >= BB * n) return;
  int b = i / n, j = i % n;
  sc[b * NP + j] = (lfe[b * NP + j] / den[b]) * ta[b * NP + j];
}

// ---------------- top-k (descending, low index wins ties) ----------------
__global__ __launch_bounds__(256) void topk_kernel(const float* __restrict__ sc,
                                                   int* __restrict__ idx, int n, int k) {
  __shared__ float sv[256]; __shared__ float rv[256]; __shared__ int ri[256];
  int b = blockIdx.x;
  sv[threadIdx.x] = (threadIdx.x < n) ? sc[b * NP + threadIdx.x] : -3.4e38f;
  __syncthreads();
  for (int r = 0; r < k; ++r) {
    rv[threadIdx.x] = sv[threadIdx.x]; ri[threadIdx.x] = threadIdx.x;
    __syncthreads();
    for (int st = 128; st > 0; st >>= 1) {
      if (threadIdx.x < st) {
        if (rv[threadIdx.x + st] > rv[threadIdx.x]) {
          rv[threadIdx.x] = rv[threadIdx.x + st]; ri[threadIdx.x] = ri[threadIdx.x + st];
        }
      }
      __syncthreads();
    }
    if (threadIdx.x == 0) { idx[b * NP + r] = ri[0]; sv[ri[0]] = -3.4e38f; }
    __syncthreads();
  }
}

__global__ void gather_kernel(const float* __restrict__ h, const int* __restrict__ idx,
                              float* __restrict__ ho, int Nin, int k, int total) {
  int i = blockIdx.x * 256 + threadIdx.x;
  if (i >= total) return;
  int c = i % DD, t = (i / DD) % (k + 1), b = i / (DD * (k + 1));
  int src = (t == 0) ? 0 : 1 + idx[b * NP + t - 1];
  ho[i] = h[((size_t)b * Nin + src) * DD + c];
}

// ---------------- host orchestration ----------------
extern "C" void kernel_launch(void* const* d_in, const int* in_sizes, int n_in,
                              void* d_out, int out_size, void* d_ws, size_t ws_size,
                              hipStream_t stream) {
  (void)in_sizes; (void)n_in; (void)out_size; (void)ws_size;
  // input order: x, patch_w, patch_b, cls, pos, 12x{ln1_g,ln1_b,qkv_w,qkv_b,proj_w,
  // proj_b,ln2_g,ln2_b,fc1_w,fc1_b,fc2_w,fc2_b}, ln_g, ln_b, head_w, head_b
  const float* x       = (const float*)d_in[0];
  const float* patch_w = (const float*)d_in[1];
  const float* patch_b = (const float*)d_in[2];
  const float* cls     = (const float*)d_in[3];
  const float* pos     = (const float*)d_in[4];
  const float* ln_g    = (const float*)d_in[149];
  const float* ln_b    = (const float*)d_in[150];
  const float* head_w  = (const float*)d_in[151];
  const float* head_b  = (const float*)d_in[152];

  char* p = (char*)d_ws;
  auto carve = [&](size_t bytes) -> char* {
    char* r = p; p += (bytes + 255) & ~(size_t)255; return r;
  };
  float*    hA    = (float*)carve((size_t)BB * NMAX * DD * 4);
  float*    hB    = (float*)carve((size_t)BB * NMAX * DD * 4);
  _Float16* lnbuf = (_Float16*)carve((size_t)BB * NMAX * DD * 2);
  float*    qkv   = (float*)carve((size_t)BB * NMAX * 3 * DD * 4);
  float*    att   = (float*)carve((size_t)BB * HH * NMAX * NSMAX * 4);
  _Float16* msaH  = (_Float16*)carve((size_t)BB * NMAX * DD * 2);
  _Float16* mlpH  = (_Float16*)carve((size_t)BB * NMAX * MLPD * 2);
  // transposed f16 weights (reused every layer)
  _Float16* qkvT  = (_Float16*)carve((size_t)DD * 3 * DD * 2);
  _Float16* projT = (_Float16*)carve((size_t)DD * DD * 2);
  _Float16* fc1T  = (_Float16*)carve((size_t)DD * MLPD * 2);
  _Float16* fc2T  = (_Float16*)carve((size_t)MLPD * DD * 2);
  _Float16* patchT= (_Float16*)carve((size_t)DD * DD * 2);
  _Float16* headT = (_Float16*)carve((size_t)DD * 1000 * 2);
  float*    cm    = (float*)carve((size_t)BB * HH * NP * 4);
  float*    c0    = (float*)carve((size_t)BB * HH * 4);
  float*    ta    = (float*)carve((size_t)BB * NP * 4);
  float*    gre   = (float*)carve(NP * 4);
  float*    gim   = (float*)carve(NP * 4);
  float*    lfe   = (float*)carve((size_t)BB * NP * 4);
  float*    den   = (float*)carve(BB * 4);
  float*    sc    = (float*)carve((size_t)BB * NP * 4);
  int*      idxv  = (int*)carve((size_t)BB * NP * 4);
  _Float16* im2   = mlpH;  // alias: im2col only used before block 0's MLP
  float*    xe    = qkv;   // alias: patch-embed out consumed before first QKV GEMM

  // ---- patch embed ----
  { int tot = DD * DD;  // patch_w is already [Dout][768] = [N][K]
    convd_kernel<<<(tot + 255) / 256, 256, 0, stream>>>(patch_w, patchT, tot); }
  { int tot = BB * NP * DD;
    im2col_kernel<<<(tot + 255) / 256, 256, 0, stream>>>(x, im2, tot); }
  gemm_kernel<<<dim3((BB * NP + 15) / 16, (DD + 63) / 64), 128, 0, stream>>>(
      im2, DD, patchT, patch_b, nullptr, 0, xe, nullptr, DD, BB * NP, DD, DD, 0);
  { int tot = BB * NMAX * DD;
    assemble_kernel<<<(tot + 255) / 256, 256, 0, stream>>>(xe, cls, pos, hA, tot); }

  static const int cfg[12] = {196, 196, 176, 176, 144, 144, 120, 120, 96, 96, 72, 72};
  float* h = hA; float* hAlt = hB;
  int N = NMAX;

  for (int L = 0; L < 12; ++L) {
    int base = 5 + L * 12;
    const float* ln1g  = (const float*)d_in[base + 0];
    const float* ln1b  = (const float*)d_in[base + 1];
    const float* qkvw  = (const float*)d_in[base + 2];
    const float* qkvb  = (const float*)d_in[base + 3];
    const float* projw = (const float*)d_in[base + 4];
    const float* projb = (const float*)d_in[base + 5];
    const float* ln2g  = (const float*)d_in[base + 6];
    const float* ln2b  = (const float*)d_in[base + 7];
    const float* fc1w  = (const float*)d_in[base + 8];
    const float* fc1b  = (const float*)d_in[base + 9];
    const float* fc2w  = (const float*)d_in[base + 10];
    const float* fc2b  = (const float*)d_in[base + 11];

    // f32 [K][N] -> f16 [N][K] weight conversion for this layer
    { int t1 = DD * 3 * DD;
      convw_kernel<<<(t1 + 255) / 256, 256, 0, stream>>>(qkvw, qkvT, DD, 3 * DD);
      int t2 = DD * DD;
      convw_kernel<<<(t2 + 255) / 256, 256, 0, stream>>>(projw, projT, DD, DD);
      int t3 = DD * MLPD;
      convw_kernel<<<(t3 + 255) / 256, 256, 0, stream>>>(fc1w, fc1T, DD, MLPD);
      convw_kernel<<<(t3 + 255) / 256, 256, 0, stream>>>(fc2w, fc2T, MLPD, DD); }

    int rows = BB * N;
    int NS = (N + 31) & ~31;
    // LN1 -> f16, QKV GEMM
    ln_kernel<<<rows, 256, 0, stream>>>(h, ln1g, ln1b, lnbuf);
    gemm_kernel<<<dim3((rows + 15) / 16, (3 * DD + 63) / 64), 128, 0, stream>>>(
        lnbuf, DD, qkvT, qkvb, nullptr, 0, qkv, nullptr, 3 * DD, rows, 3 * DD, DD, 0);
    // attention
    qk_kernel<<<dim3((N + 15) / 16, (N + 63) / 64, BB * HH), 128, 0, stream>>>(qkv, att, N, NS);
    softmax_kernel<<<BB * HH * N, 256, 0, stream>>>(att, N, NS);
    av_kernel<<<dim3((N + 15) / 16, 1, BB * HH), 128, 0, stream>>>(att, qkv, msaH, N, NS);
    // proj + residual (in place into h)
    gemm_kernel<<<dim3((rows + 15) / 16, (DD + 63) / 64), 128, 0, stream>>>(
        msaH, DD, projT, projb, h, DD, h, nullptr, DD, rows, DD, DD, 0);
    // token selector
    if (cfg[L] != N - 1) {
      int n = N - 1, k = cfg[L];
      colmean_kernel<<<BB * HH, 256, 0, stream>>>(att, cm, c0, N, NS);
      tokenattn_kernel<<<BB, 256, 0, stream>>>(att, cm, c0, ta, N, NS);
      gfilter_kernel<<<1, 256, 0, stream>>>(gre, gim, n, 0.15f);
      lfe_kernel<<<BB * n, 256, 0, stream>>>(h, gre, gim, lfe, N);
      denom_kernel<<<BB, 256, 0, stream>>>(h, den, N);
      { int tot = BB * n;
        scores_kernel<<<(tot + 255) / 256, 256, 0, stream>>>(lfe, den, ta, sc, n); }
      topk_kernel<<<BB, 256, 0, stream>>>(sc, idxv, n, k);
      { int tot = BB * (k + 1) * DD;
        gather_kernel<<<(tot + 255) / 256, 256, 0, stream>>>(h, idxv, hAlt, N, k, tot); }
      float* t = h; h = hAlt; hAlt = t;
      N = k + 1;
      rows = BB * N;
    }
    // MLP: LN2 -> fc1(GELU, f16 out) -> fc2 + residual
    ln_kernel<<<rows, 256, 0, stream>>>(h, ln2g, ln2b, lnbuf);
    gemm_kernel<<<dim3((rows + 15) / 16, (MLPD + 63) / 64), 128, 0, stream>>>(
        lnbuf, DD, fc1T, fc1b, nullptr, 0, nullptr, mlpH, MLPD, rows, MLPD, DD, /*gelu=*/1);
    gemm_kernel<<<dim3((rows + 15) / 16, (DD + 63) / 64), 128, 0, stream>>>(
        mlpH, MLPD, fc2T, fc2b, h, DD, h, nullptr, DD, rows, DD, MLPD, 0);
  }

  // final LN + head on CLS rows (lda trick: row b -> lnbuf[b*N*DD])
  { int th = DD * 1000;
    convw_kernel<<<(th + 255) / 256, 256, 0, stream>>>(head_w, headT, DD, 1000); }
  ln_kernel<<<BB * N, 256, 0, stream>>>(h, ln_g, ln_b, lnbuf);
  gemm_kernel<<<dim3((BB + 15) / 16, (1000 + 63) / 64), 128, 0, stream>>>(
      lnbuf, N * DD, headT, head_b, nullptr, 0,
      (float*)d_out, nullptr, 1000, BB, 1000, DD, 0);
}
